// MultiHeadAttention_71871982731819
// MI455X (gfx1250) — matrix operand order repaired and checked by reading, not compiled, not run on hardware
//
#include <hip/hip_runtime.h>

typedef unsigned short u16;
typedef unsigned int   u32;
typedef __attribute__((ext_vector_type(16))) __bf16 bf16x16;
typedef __attribute__((ext_vector_type(8)))  float  floatx8;

struct __align__(16) U4 { u32 x, y, z, w; };

union Frag16 {
  bf16x16 v;
  U4      q[2];
  u16     e[16];
};

__device__ __forceinline__ u16 f32_to_bf16(float f) {
  u32 u = __builtin_bit_cast(u32, f);
  u32 r = u + 0x7FFFu + ((u >> 16) & 1u);   // round-to-nearest-even
  return (u16)(r >> 16);
}

// ---------------------------------------------------------------------------
// CDNA5 async global->LDS copy (no VGPR intermediary, tracked by ASYNCcnt).
// DS offset of a __shared__ object == low 32 bits of its flat address
// (ISA 10.2: LDS aperture decode).
// ---------------------------------------------------------------------------
__device__ __forceinline__ void async_ld_b128(void* lds_ptr, const void* gptr) {
  u32 l = (u32)(uintptr_t)lds_ptr;
  asm volatile("global_load_async_to_lds_b128 %0, %1, off"
               :: "v"(l), "v"(gptr)
               : "memory");
}
__device__ __forceinline__ void wait_async0() {
  asm volatile("s_wait_asynccnt 0" ::: "memory");
}

// ---------------------------------------------------------------------------
// fp32 -> bf16 bulk convert (8 elements / thread)
// ---------------------------------------------------------------------------
__global__ __launch_bounds__(256) void cvt_f32_bf16_kernel(
    const float* __restrict__ src, u16* __restrict__ dst, int n8) {
  int i = blockIdx.x * 256 + threadIdx.x;
  if (i >= n8) return;
  const float* p = src + (size_t)i * 8;
  float4 a = *(const float4*)p;
  float4 b = *(const float4*)(p + 4);
  U4 o;
  o.x = (u32)f32_to_bf16(a.x) | ((u32)f32_to_bf16(a.y) << 16);
  o.y = (u32)f32_to_bf16(a.z) | ((u32)f32_to_bf16(a.w) << 16);
  o.z = (u32)f32_to_bf16(b.x) | ((u32)f32_to_bf16(b.y) << 16);
  o.w = (u32)f32_to_bf16(b.z) | ((u32)f32_to_bf16(b.w) << 16);
  *(U4*)(dst + (size_t)i * 8) = o;
}

// ---------------------------------------------------------------------------
// fp32 W[K][N] -> bf16 Wt[N][K]  (transpose once, via 64x64 LDS tile)
// ---------------------------------------------------------------------------
__global__ __launch_bounds__(256) void cvt_transpose_kernel(
    const float* __restrict__ W, u16* __restrict__ Wt, int N, int K) {
  __shared__ __align__(16) u16 sT[64][72];
  const int tid = threadIdx.x;
  const int n0 = blockIdx.x * 64;
  const int k0 = blockIdx.y * 64;

  const int kk = tid >> 2;
  const int nb = (tid & 3) * 16;
  const float* p = W + (size_t)(k0 + kk) * N + n0 + nb;
#pragma unroll
  for (int c = 0; c < 4; ++c) {
    float4 f = *(const float4*)(p + c * 4);
    sT[nb + c * 4 + 0][kk] = f32_to_bf16(f.x);
    sT[nb + c * 4 + 1][kk] = f32_to_bf16(f.y);
    sT[nb + c * 4 + 2][kk] = f32_to_bf16(f.z);
    sT[nb + c * 4 + 3][kk] = f32_to_bf16(f.w);
  }
  __syncthreads();

  const int nn = tid >> 2;
  const int kb = (tid & 3) * 16;
  u16* q = Wt + (size_t)(n0 + nn) * K + k0 + kb;
  *(U4*)q       = *(const U4*)&sT[nn][kb];
  *(U4*)(q + 8) = *(const U4*)&sT[nn][kb + 8];
}

// ---------------------------------------------------------------------------
// Tiled bf16 GEMM: C[M,N] = (A[M,K] * Bt[N,K]^T + bias) * oscale
// Bt = pre-transposed weight. 256 threads (8 wave32), 128x128 tile, BK=32.
// Double-buffered LDS filled by ASYNC global->LDS loads (zero staging VGPRs,
// no spills); one s_wait_asynccnt + one barrier per k-step.
// Waves 2(M)x4(N), each 64x32 = 4x2 WMMA tiles.
// ---------------------------------------------------------------------------
#define BKK 32

__global__ __launch_bounds__(256) void gemm_bf16_kernel(
    const u16* __restrict__ A, const u16* __restrict__ Bt,
    const float* __restrict__ bias, float oscale,
    u16* __restrict__ Cb, float* __restrict__ Cf,
    int M, int N, int K) {
  __shared__ __align__(16) u16 sA[2][128][40];   // [buf][m][k]
  __shared__ __align__(16) u16 sB[2][128][40];   // [buf][n][k]

  const int tid = threadIdx.x;
  const int lane = tid & 31, wave = tid >> 5;
  const int half = lane >> 4, lm = lane & 15;
  const int wm = wave >> 2, wn = wave & 3;       // 2 x 4 wave grid
  const int mBase = blockIdx.y * 128;
  const int nBase = blockIdx.x * 128;

  const floatx8 zero8 = {0.f, 0.f, 0.f, 0.f, 0.f, 0.f, 0.f, 0.f};
  floatx8 acc[4][2];
#pragma unroll
  for (int i = 0; i < 4; ++i)
#pragma unroll
    for (int j = 0; j < 2; ++j) acc[i][j] = zero8;

  // staging: half a row of each tile per thread, 2x async b128 each
  const int sr = tid >> 1;            // row 0..127
  const int sc = (tid & 1) * 16;      // col base 0 / 16
  const u16* pa = A  + (size_t)(mBase + sr) * K + sc;
  const u16* pb = Bt + (size_t)(nBase + sr) * K + sc;

  // prologue: async-stage tile 0 into buffer 0
  async_ld_b128(&sA[0][sr][sc],     pa);
  async_ld_b128(&sA[0][sr][sc + 8], pa + 8);
  async_ld_b128(&sB[0][sr][sc],     pb);
  async_ld_b128(&sB[0][sr][sc + 8], pb + 8);

  const int nIter = K / BKK;
  for (int it = 0; it < nIter; ++it) {
    wait_async0();        // our async fills of the buffer everyone reads next
    __syncthreads();
    const int buf = it & 1;
    if (it + 1 < nIter) {
      const u16* qa = pa + (it + 1) * BKK;
      const u16* qb = pb + (it + 1) * BKK;
      async_ld_b128(&sA[buf ^ 1][sr][sc],     qa);
      async_ld_b128(&sA[buf ^ 1][sr][sc + 8], qa + 8);
      async_ld_b128(&sB[buf ^ 1][sr][sc],     qb);
      async_ld_b128(&sB[buf ^ 1][sr][sc + 8], qb + 8);
      if (it + 2 < nIter) {
        __builtin_prefetch(pa + (it + 2) * BKK, 0, 1);
        __builtin_prefetch(pb + (it + 2) * BKK, 0, 1);
      }
    }

    Frag16 af[4], bfr[2];
#pragma unroll
    for (int i = 0; i < 4; ++i) {
      const u16* p = &sA[buf][wm * 64 + i * 16 + lm][0];
      af[i].q[0] = *(const U4*)(p + 8 * half);        // K = e + 8*half
      af[i].q[1] = *(const U4*)(p + 16 + 8 * half);   // K = e+8 + 8*half
    }
#pragma unroll
    for (int j = 0; j < 2; ++j) {
      const u16* p = &sB[buf][wn * 32 + j * 16 + lm][0];
      bfr[j].q[0] = *(const U4*)(p + 8 * half);       // K = e + 8*half
      bfr[j].q[1] = *(const U4*)(p + 16 + 8 * half);
    }
#pragma unroll
    for (int i = 0; i < 4; ++i)
#pragma unroll
      for (int j = 0; j < 2; ++j)
        acc[i][j] = __builtin_amdgcn_wmma_f32_16x16x32_bf16(
            false, af[i].v, false, bfr[j].v, (short)0, acc[i][j], false, false);
  }

  // epilogue: C layout N = lm, M = v + 8*half
#pragma unroll
  for (int j = 0; j < 2; ++j) {
    const int col = nBase + wn * 32 + j * 16 + lm;
    const float bb = bias ? bias[col] : 0.f;
#pragma unroll
    for (int i = 0; i < 4; ++i) {
#pragma unroll
      for (int v = 0; v < 8; ++v) {
        const int row = mBase + wm * 64 + i * 16 + v + 8 * half;
        const float val = (acc[i][j][v] + bb) * oscale;
        if (Cf) Cf[(size_t)row * N + col] = val;
        else    Cb[(size_t)row * N + col] = f32_to_bf16(val);
      }
    }
  }
}

// ---------------------------------------------------------------------------
// Flash attention. One block = (head, 128 q rows); 8 waves x 16 q rows.
// 64 keys/iter. K staged by ASYNC global->LDS; V^T staged via short-lived
// register scatter at the top of each iteration. Double-buffered, 1 barrier.
// Q pre-scaled by 1/sqrt(Hd) in the Q-projection epilogue.
// ---------------------------------------------------------------------------
#define HEADS 16
#define HD 64
#define DMODEL 1024
#define QB 128
#define KBLK 64

__global__ __launch_bounds__(256) void flash_attn_kernel(
    const u16* __restrict__ Q, const u16* __restrict__ Km,
    const u16* __restrict__ V, u16* __restrict__ O, int S) {
  __shared__ __align__(16) u16 sK[2][KBLK][72];    // [buf][key][d]
  __shared__ __align__(16) u16 sVt[2][HD][72];     // [buf][d][key]
  __shared__ __align__(16) u16 sP[8][16][72];      // per-wave P [qrow][key]

  const int tid = threadIdx.x;
  const int lane = tid & 31, wave = tid >> 5;
  const int half = lane >> 4, lm = lane & 15;
  const int h = blockIdx.y;
  const int q0 = blockIdx.x * QB;
  const int hc = h * HD;
  const int qr = q0 + wave * 16 + lm;

  // Q A-fragments (fixed rows): 16x64 as 2 x (16x32)
  Frag16 aq[2];
#pragma unroll
  for (int ds = 0; ds < 2; ++ds) {
    const u16* p = Q + (size_t)qr * DMODEL + hc + ds * 32;
    aq[ds].q[0] = *(const U4*)(p + 8 * half);
    aq[ds].q[1] = *(const U4*)(p + 16 + 8 * half);
  }

  const floatx8 zero8 = {0.f, 0.f, 0.f, 0.f, 0.f, 0.f, 0.f, 0.f};
  floatx8 o[4];
#pragma unroll
  for (int dt = 0; dt < 4; ++dt) o[dt] = zero8;
  float mrow[8], lrow[8];
#pragma unroll
  for (int v = 0; v < 8; ++v) { mrow[v] = -1e30f; lrow[v] = 0.f; }

  const int vk = tid >> 2;          // key row 0..63 for staging
  const int vd = (tid & 3) * 16;    // d base for staging

  auto stageK = [&](int kb, int buf) {
    const u16* pk = Km + (size_t)(kb + vk) * DMODEL + hc + vd;
    async_ld_b128(&sK[buf][vk][vd],     pk);
    async_ld_b128(&sK[buf][vk][vd + 8], pk + 8);
  };
  auto stageV = [&](int kb, int buf) {   // transpose scatter, short-lived regs
    const u16* pv = V + (size_t)(kb + vk) * DMODEL + hc + vd;
    Frag16 t;
    t.q[0] = *(const U4*)pv;
    t.q[1] = *(const U4*)(pv + 8);
#pragma unroll
    for (int j = 0; j < 16; ++j) sVt[buf][vd + j][vk] = t.e[j];
  };

  stageK(0, 0);
  stageV(0, 0);

  const int nIt = S / KBLK;
  for (int it = 0; it < nIt; ++it) {
    wait_async0();
    __syncthreads();
    const int buf = it & 1;
    if (it + 1 < nIt) {
      stageK((it + 1) * KBLK, buf ^ 1);
      stageV((it + 1) * KBLK, buf ^ 1);
    }

    // ---- S = Q @ K^T : 4 key-tiles, contraction over d (2x32) ----
    floatx8 s[4];
#pragma unroll
    for (int kt = 0; kt < 4; ++kt) s[kt] = zero8;
#pragma unroll
    for (int kt = 0; kt < 4; ++kt) {
#pragma unroll
      for (int ds = 0; ds < 2; ++ds) {
        Frag16 bk;  // B frag: N = key (lm), K = d = ds*32 + (e<8?e:e+8) + 8*half
        const u16* p = &sK[buf][kt * 16 + lm][ds * 32 + 8 * half];
        bk.q[0] = *(const U4*)p;
        bk.q[1] = *(const U4*)(p + 16);
        s[kt] = __builtin_amdgcn_wmma_f32_16x16x32_bf16(
            false, aq[ds].v, false, bk.v, (short)0, s[kt], false, false);
      }
    }

    // ---- online softmax over the 64 new keys ----
#pragma unroll
    for (int v = 0; v < 8; ++v) {
      float x0 = s[0][v], x1 = s[1][v], x2 = s[2][v], x3 = s[3][v];
      float mx = fmaxf(fmaxf(x0, x1), fmaxf(x2, x3));
#pragma unroll
      for (int msk = 1; msk < 16; msk <<= 1)
        mx = fmaxf(mx, __shfl_xor(mx, msk, 32));
      const float nm = fmaxf(mrow[v], mx);
      const float p0 = __expf(x0 - nm), p1 = __expf(x1 - nm);
      const float p2 = __expf(x2 - nm), p3 = __expf(x3 - nm);
      float sum = p0 + p1 + p2 + p3;
#pragma unroll
      for (int msk = 1; msk < 16; msk <<= 1) sum += __shfl_xor(sum, msk, 32);
      const float alpha = __expf(mrow[v] - nm);
      lrow[v] = lrow[v] * alpha + sum;
      mrow[v] = nm;
      o[0][v] *= alpha; o[1][v] *= alpha; o[2][v] *= alpha; o[3][v] *= alpha;
      const int r = v + 8 * half;
      sP[wave][r][ 0 + lm] = f32_to_bf16(p0);
      sP[wave][r][16 + lm] = f32_to_bf16(p1);
      sP[wave][r][32 + lm] = f32_to_bf16(p2);
      sP[wave][r][48 + lm] = f32_to_bf16(p3);
    }

    // ---- O += P @ V : contraction over 64 keys (2x32), 4 d-tiles ----
#pragma unroll
    for (int ds = 0; ds < 2; ++ds) {
      Frag16 ap;  // A frag of P: row = lm, K = ds*32 + (e<8?e:e+8) + 8*half
      const u16* p = &sP[wave][lm][ds * 32];
      ap.q[0] = *(const U4*)(p + 8 * half);
      ap.q[1] = *(const U4*)(p + 16 + 8 * half);
#pragma unroll
      for (int dt = 0; dt < 4; ++dt) {
        Frag16 bv;  // B frag of V: N = d (lm), K = key = ds*32 + (e<8?e:e+8) + 8*half
        const u16* pb = &sVt[buf][dt * 16 + lm][ds * 32 + 8 * half];
        bv.q[0] = *(const U4*)pb;
        bv.q[1] = *(const U4*)(pb + 16);
        o[dt] = __builtin_amdgcn_wmma_f32_16x16x32_bf16(
            false, ap.v, false, bv.v, (short)0, o[dt], false, false);
      }
    }
  }

  // ---- normalize and store attention output (bf16, [S, H*Hd]) ----
#pragma unroll
  for (int v = 0; v < 8; ++v) {
    const float inv = 1.f / lrow[v];
    const int row = q0 + wave * 16 + v + 8 * half;
#pragma unroll
    for (int dt = 0; dt < 4; ++dt) {
      O[(size_t)row * DMODEL + hc + dt * 16 + lm] =
          f32_to_bf16(o[dt][v] * inv);
    }
  }
}

// ---------------------------------------------------------------------------
// Host-side orchestration
// ---------------------------------------------------------------------------
extern "C" void kernel_launch(void* const* d_in, const int* in_sizes, int n_in,
                              void* d_out, int out_size, void* d_ws,
                              size_t ws_size, hipStream_t stream) {
  const float* x  = (const float*)d_in[0];
  const float* wq = (const float*)d_in[1];
  const float* bq = (const float*)d_in[2];
  const float* wk = (const float*)d_in[3];
  const float* bk = (const float*)d_in[4];
  const float* wv = (const float*)d_in[5];
  const float* bv = (const float*)d_in[6];
  const float* wo = (const float*)d_in[7];
  const float* bo = (const float*)d_in[8];

  const int S = 4096, D = 1024;
  char* w = (char*)d_ws;
  const size_t MB = 1024 * 1024;
  u16* xb  = (u16*)(w + 0 * MB);    // 8 MB  x   (bf16)
  u16* wqt = (u16*)(w + 8 * MB);    // 2 MB  wq^T
  u16* wkt = (u16*)(w + 10 * MB);   // 2 MB  wk^T
  u16* wvt = (u16*)(w + 12 * MB);   // 2 MB  wv^T
  u16* wot = (u16*)(w + 14 * MB);   // 2 MB  wo^T
  u16* Qb  = (u16*)(w + 16 * MB);   // 8 MB
  u16* Kb  = (u16*)(w + 24 * MB);   // 8 MB
  u16* Vb  = (u16*)(w + 32 * MB);   // 8 MB
  u16* AOb = (u16*)(w + 40 * MB);   // 8 MB

  const int nx = S * D;   // 4194304
  cvt_f32_bf16_kernel<<<nx / 8 / 256, 256, 0, stream>>>(x, xb, nx / 8);

  dim3 tg(D / 64, D / 64);  // (16,16)
  cvt_transpose_kernel<<<tg, 256, 0, stream>>>(wq, wqt, D, D);
  cvt_transpose_kernel<<<tg, 256, 0, stream>>>(wk, wkt, D, D);
  cvt_transpose_kernel<<<tg, 256, 0, stream>>>(wv, wvt, D, D);
  cvt_transpose_kernel<<<tg, 256, 0, stream>>>(wo, wot, D, D);

  const float qscale = 0.125f;  // 1/sqrt(64), folded into Q projection
  dim3 gg(D / 128, S / 128);    // (8, 32)
  gemm_bf16_kernel<<<gg, 256, 0, stream>>>(xb, wqt, bq, qscale, Qb, nullptr, S, D, D);
  gemm_bf16_kernel<<<gg, 256, 0, stream>>>(xb, wkt, bk, 1.0f,  Kb, nullptr, S, D, D);
  gemm_bf16_kernel<<<gg, 256, 0, stream>>>(xb, wvt, bv, 1.0f,  Vb, nullptr, S, D, D);

  flash_attn_kernel<<<dim3(S / QB, HEADS), 256, 0, stream>>>(Qb, Kb, Vb, AOb, S);

  gemm_bf16_kernel<<<gg, 256, 0, stream>>>(AOb, wot, bo, 1.0f, nullptr,
                                           (float*)d_out, S, D, D);
}
